// VanillaRNN_82068235092128
// MI455X (gfx1250) — matrix operand order repaired and checked by reading, not compiled, not run on hardware
//
#include <hip/hip_runtime.h>
#include <hip/hip_bf16.h>

#define Bb 64
#define Tt 2048
#define Hh 512
#define NWG 16
#define NTHREADS 256

typedef __bf16 bf16_t;
typedef __attribute__((ext_vector_type(16))) __bf16 v16bf;
typedef __attribute__((ext_vector_type(8)))  __bf16 v8bf;
typedef __attribute__((ext_vector_type(8)))  float  v8f;

__device__ __forceinline__ v16bf cat16(v8bf lo, v8bf hi) {
  return __builtin_shufflevector(lo, hi, 0,1,2,3,4,5,6,7,8,9,10,11,12,13,14,15);
}

// float -> bf16, round-to-nearest-even (bit ops: guaranteed to lower cleanly)
__device__ __forceinline__ bf16_t f2bf(float f) {
  unsigned u = __builtin_bit_cast(unsigned, f);
  u += 0x7fffu + ((u >> 16) & 1u);
  unsigned short s = (unsigned short)(u >> 16);
  return __builtin_bit_cast(bf16_t, s);
}

// Init: zero the grid-barrier counter, convert W (f32 [H][H]) and h0 (f32 [B][H]) to bf16.
__global__ void rnn_init_kernel(const float* __restrict__ W,
                                const float* __restrict__ hidden,
                                bf16_t* __restrict__ Wbf,
                                bf16_t* __restrict__ h0,
                                int* __restrict__ counter) {
  int i = blockIdx.x * blockDim.x + threadIdx.x;
  if (i == 0) *counter = 0;
  int stride = gridDim.x * blockDim.x;
  for (int j = i; j < Hh * Hh; j += stride) Wbf[j] = f2bf(W[j]);
  for (int j = i; j < Bb * Hh; j += stride) h0[j]  = f2bf(hidden[j]);
}

// Persistent RNN kernel: 16 WGs x 8 wave32 = 128 waves = 4 M-tiles x 32 N-tiles.
// Each wave owns one 16x16 output tile; W fragments (16 k-steps) live in VGPRs
// for the whole T loop. Device-wide atomic barrier per step.
__launch_bounds__(NTHREADS, 1)
__global__ void rnn_persistent_kernel(const float* __restrict__ x,     // [B,T]
                                      const float* __restrict__ W0,    // [H]
                                      const float* __restrict__ b0,    // [H]
                                      const float* __restrict__ bvec,  // [H]
                                      const bf16_t* __restrict__ Wbf,  // [H,H] bf16
                                      bf16_t* __restrict__ hbuf0,      // [B,H] bf16
                                      bf16_t* __restrict__ hbuf1,      // [B,H] bf16
                                      int* __restrict__ counter,
                                      float* __restrict__ yout,        // [B,T,H]
                                      float* __restrict__ hout) {      // [B,H]
  const int tid    = threadIdx.x;
  const int lane   = tid & 31;
  const int wave   = tid >> 5;            // 0..7
  const int lane15 = lane & 15;
  const int hi16   = lane >> 4;           // 0 for lanes 0-15, 1 for 16-31

  const int mtile = wave & 3;                          // 0..3   (M = 64 / 16)
  const int ntile = (blockIdx.x << 1) | (wave >> 2);   // 0..31  (N = 512 / 16)
  const int mb = mtile << 4;
  const int nb = ntile << 4;
  const int col = nb + lane15;            // output column owned by this lane

  const float w0n = W0[col];
  const float bsn = b0[col] + bvec[col];

  // ---- Preload B operand: W^T column tile, full K chain, register-resident.
  // B 32x16 bf16, this code's k order (same as A): lanes 0-15 hold K=0..15 of col N=lane, lanes 16-31 K=16..31.
  // B[k][n] = W[n][k]  -> 16 contiguous bf16 from row `col` of W.
  v16bf wfrag[16];
  const int koffB = hi16 << 4;            // 0 or 16
#pragma unroll
  for (int k = 0; k < 16; ++k) {
    const bf16_t* wp = Wbf + (size_t)col * Hh + k * 32 + koffB;
    v8bf lo = *(const v8bf*)(wp);
    v8bf hi = *(const v8bf*)(wp + 8);
    wfrag[k] = cat16(lo, hi);
  }

  // A 16x32 bf16, this code's k order (same as B): M = lane15; lanes 0-15 take K {0..15},
  // lanes 16-31 take K {16..31} (two b128 loads per lane).
  const int arow  = mb + lane15;
  const int koffA = hi16 << 4;            // 0 or 16

  for (int t = 0; t < Tt; ++t) {
    const bf16_t* __restrict__ hread = (t & 1) ? hbuf1 : hbuf0;
    bf16_t* __restrict__ hwrite      = (t & 1) ? hbuf0 : hbuf1;

    v8f acc0 = {0.f,0.f,0.f,0.f,0.f,0.f,0.f,0.f};
    v8f acc1 = {0.f,0.f,0.f,0.f,0.f,0.f,0.f,0.f};
    const bf16_t* ap = hread + (size_t)arow * Hh + koffA;

#pragma unroll
    for (int k = 0; k < 16; k += 2) {   // two independent accumulator chains
      v8bf lo0 = *(const v8bf*)(ap + k * 32);
      v8bf hi0 = *(const v8bf*)(ap + k * 32 + 8);
      v16bf a0 = cat16(lo0, hi0);
      acc0 = __builtin_amdgcn_wmma_f32_16x16x32_bf16(
                 false, a0, false, wfrag[k], (short)0, acc0, false, false);

      v8bf lo1 = *(const v8bf*)(ap + (k + 1) * 32);
      v8bf hi1 = *(const v8bf*)(ap + (k + 1) * 32 + 8);
      v16bf a1 = cat16(lo1, hi1);
      acc1 = __builtin_amdgcn_wmma_f32_16x16x32_bf16(
                 false, a1, false, wfrag[k + 1], (short)0, acc1, false, false);
    }
    v8f acc = acc0 + acc1;

    // C/D 16x16 f32 layout: VGPR r -> row mb + r + 8*hi16, col = nb + lane15.
#pragma unroll
    for (int r = 0; r < 8; ++r) {
      const int row = mb + r + (hi16 << 3);
      const float xv = x[row * Tt + t];                 // input scalar x[b][t]
      const float v  = tanhf(acc[r] + xv * w0n + bsn);
      __builtin_nontemporal_store(v, yout + ((size_t)row * Tt + t) * Hh + col);
      hwrite[row * Hh + col] = f2bf(v);
      if (t == Tt - 1) hout[row * Hh + col] = v;
    }

    // Device-wide barrier: make h_{t+1} visible, then all 16 WGs rendezvous.
    if (t < Tt - 1) {
      __threadfence();
      __syncthreads();
      if (tid == 0) {
        __hip_atomic_fetch_add(counter, 1, __ATOMIC_RELEASE, __HIP_MEMORY_SCOPE_AGENT);
        const int target = NWG * (t + 1);
        while (__hip_atomic_load(counter, __ATOMIC_ACQUIRE, __HIP_MEMORY_SCOPE_AGENT) < target) {
          __builtin_amdgcn_s_sleep(2);
        }
      }
      __syncthreads();
    }
  }
}

extern "C" void kernel_launch(void* const* d_in, const int* in_sizes, int n_in,
                              void* d_out, int out_size, void* d_ws, size_t ws_size,
                              hipStream_t stream) {
  const float* x      = (const float*)d_in[0];  // [B,T,1]
  const float* hidden = (const float*)d_in[1];  // [1,B,H]
  const float* W0     = (const float*)d_in[2];  // [H,1]
  const float* b0     = (const float*)d_in[3];  // [H]
  const float* W      = (const float*)d_in[4];  // [H,H]
  const float* bvec   = (const float*)d_in[5];  // [H]

  float* yout = (float*)d_out;                          // [B,T,H]
  float* hout = yout + (size_t)Bb * Tt * Hh;            // [1,B,H]

  char* ws = (char*)d_ws;
  int*    counter = (int*)ws;                                   // barrier counter
  bf16_t* Wbf     = (bf16_t*)(ws + 256);                        // 512 KB
  bf16_t* h0      = (bf16_t*)(ws + 256 + (size_t)Hh * Hh * 2);  // 64 KB
  bf16_t* h1      = h0 + (size_t)Bb * Hh;                       // 64 KB

  rnn_init_kernel<<<256, 256, 0, stream>>>(W, hidden, Wbf, h0, counter);
  rnn_persistent_kernel<<<NWG, NTHREADS, 0, stream>>>(
      x, W0, b0, bvec, Wbf, h0, h1, counter, yout, hout);
}
